// ConcatNotEqualSelfAtt2_55276229100119
// MI455X (gfx1250) — matrix operand order repaired and checked by reading, not compiled, not run on hardware
//
#include <hip/hip_runtime.h>

typedef __attribute__((ext_vector_type(16))) __bf16 v16bf;
typedef __attribute__((ext_vector_type(8)))  float  v8f;
typedef unsigned short u16s;

union ABu { v16bf v; unsigned int u[8]; };

__device__ __forceinline__ v8f vzero8() {
  v8f z = {0.f,0.f,0.f,0.f,0.f,0.f,0.f,0.f};
  return z;
}

__device__ __forceinline__ unsigned int pack_bf16x2(float a, float b) {
  unsigned int ua = __builtin_bit_cast(unsigned int, a);
  unsigned int ub = __builtin_bit_cast(unsigned int, b);
  ua = (ua + 0x7FFFu + ((ua >> 16) & 1u)) >> 16;   // round-to-nearest-even
  ub = (ub + 0x7FFFu + ((ub >> 16) & 1u)) >> 16;
  return ua | (ub << 16);
}

enum { Bn = 2048, Ln = 30, Dn = 512, Xn = 512, Hn = 4, OUTn = 512, ITERn = 4 };
enum { KC = 512 };   // K depth of all GEMMs (compile-time -> fully unrolled k-loop)

// ---------------------------------------------------------------------------
// f32 -> bf16 elementwise conversion (8 elements / thread)
// ---------------------------------------------------------------------------
__global__ __launch_bounds__(256) void cvt_f32_bf16(
    const float* __restrict__ in, u16s* __restrict__ out, int n8)
{
  const int i = blockIdx.x * 256 + threadIdx.x;
  if (i >= n8) return;
  const float4* p = reinterpret_cast<const float4*>(in) + (size_t)i * 2;
  const float4 a = p[0], b = p[1];
  uint4 o;
  o.x = pack_bf16x2(a.x, a.y);
  o.y = pack_bf16x2(a.z, a.w);
  o.z = pack_bf16x2(b.x, b.y);
  o.w = pack_bf16x2(b.z, b.w);
  reinterpret_cast<uint4*>(out)[i] = o;
}

// ---------------------------------------------------------------------------
// C[M,N] = A[M,K] @ W[N,K]^T, A/W bf16 row-major, C f32. Compile-time strides
// so every fragment load / C store is (lane base pointer + immediate offset).
// 128 threads = 4 waves as 2x2; block tile 128x64; wave tile 64x32
// (4 m-tiles x 2 n-tiles = 8 WMMA accumulators). No LDS: fragments are
// contiguous 16B runs -> direct global_load_b128.
// ---------------------------------------------------------------------------
template<int LDA, int LDW, int LDC>
__global__ __launch_bounds__(128) void gemm128x64_bf16(
    const u16s* __restrict__ A,
    const u16s* __restrict__ W,   // already offset by koff
    float* __restrict__ C)
{
  const int t    = threadIdx.x;
  const int lane = t & 31;
  const int wid  = t >> 5;
  const int hi   = lane >> 4;
  const int lm   = lane & 15;
  const int wm   = wid >> 1, wn = wid & 1;
  const int mbase = blockIdx.x * 128 + wm * 64;
  const int nbase = blockIdx.y * 64 + wn * 32;

  v8f acc[4][2];
  #pragma unroll
  for (int mt = 0; mt < 4; ++mt)
    #pragma unroll
    for (int nt = 0; nt < 2; ++nt) acc[mt][nt] = vzero8();

  // Single per-lane base pointer per stream; all tile/k offsets are immediates.
  const u16s* abase = A + (size_t)(mbase + lm) * LDA + hi * 8;
  const u16s* bbase = W + (size_t)(nbase + lm) * LDW + hi * 8;

  #pragma unroll
  for (int ks = 0; ks < KC / 32; ++ks) {
    const int k0 = ks * 32;
    ABu aU[4], bU[2];
    #pragma unroll
    for (int mt = 0; mt < 4; ++mt) {
      const u16s* ap = abase + mt * 16 * LDA + k0;
      *reinterpret_cast<uint4*>(&aU[mt].u[0]) =
          *reinterpret_cast<const uint4*>(ap);          // K = k0+hi*8 ..+7
      *reinterpret_cast<uint4*>(&aU[mt].u[4]) =
          *reinterpret_cast<const uint4*>(ap + 16);     // K = k0+16+hi*8 ..+7
    }
    #pragma unroll
    for (int nt = 0; nt < 2; ++nt) {
      const u16s* bp = bbase + nt * 16 * LDW + k0;      // K = k0+hi*8 ..+7, k0+16+hi*8 ..+7
      *reinterpret_cast<uint4*>(&bU[nt].u[0]) =
          *reinterpret_cast<const uint4*>(bp);
      *reinterpret_cast<uint4*>(&bU[nt].u[4]) =
          *reinterpret_cast<const uint4*>(bp + 16);
    }
    #pragma unroll
    for (int mt = 0; mt < 4; ++mt)
      #pragma unroll
      for (int nt = 0; nt < 2; ++nt)
        acc[mt][nt] = __builtin_amdgcn_wmma_f32_16x16x32_bf16(
            false, aU[mt].v, false, bU[nt].v, (short)0, acc[mt][nt], false, false);
  }

  // Epilogue: one base pointer; all 64 store offsets are compile-time.
  float* cbase = C + (size_t)(mbase + hi * 8) * LDC + nbase + lm;
  #pragma unroll
  for (int mt = 0; mt < 4; ++mt)
    #pragma unroll
    for (int nt = 0; nt < 2; ++nt)
      #pragma unroll
      for (int r = 0; r < 8; ++r)
        cbase[(mt * 16 + r) * LDC + nt * 16] = acc[mt][nt][r];
}

// ---------------------------------------------------------------------------
// Per-batch fused kernel: Gram (WMMA bf16, direct global fragments)
// -> EvdCred iterations -> evd_cred -> tanh stage -> lin/softmax -> AOA GLU
// -> attended. 1 block (128 threads) per batch.
// ---------------------------------------------------------------------------
__global__ __launch_bounds__(128) void fused_evd_attn(
    const float* __restrict__ right,
    const u16s*  __restrict__ right_bf,     // (B*L + 2 pad rows) x 512 bf16
    const int*   __restrict__ mask,
    const float* __restrict__ W_o,
    const float* __restrict__ b_o,
    const float* __restrict__ scales_blk,
    const float* __restrict__ W_c,
    const float* __restrict__ b_c,
    const float* __restrict__ scale_fin,
    const float* __restrict__ W2,
    const float* __restrict__ W_aoa,
    const float* __restrict__ b_aoa,
    const float* __restrict__ leftW,
    const float* __restrict__ rightW,
    float* __restrict__ out0,
    float* __restrict__ out2)
{
  __shared__ float Gs[32 * 32];
  __shared__ float Tl[Ln * OUTn];
  __shared__ float cred[32], Xb[32];
  __shared__ float evd[Ln * Hn], lins[Ln * Hn], aw[Ln * Hn], aw2s[Ln * Hn];
  __shared__ float s_valid;

  const int b    = blockIdx.x;
  const int t    = threadIdx.x;
  const int lane = t & 31;
  const int wvid = t >> 5;
  const int hi   = lane >> 4;
  const int lm   = lane & 15;
  const float* rb = right + (size_t)b * Ln * Dn;

  if (t == 0) {
    int s = 0;
    for (int l = 0; l < Ln; ++l) s += mask[b * Ln + l];
    s_valid = (float)s;
  }
  if (t < 32) Xb[t] = 1.0f;

  // ---- Gram G = R @ R^T (rows 30/31 read pad garbage; never consumed) ----
  {
    const u16s* rbb = right_bf + (size_t)b * Ln * Dn;
    const int mt = wvid >> 1, nt = wvid & 1;
    const u16s* ar = rbb + (size_t)(mt * 16 + lm) * Dn + hi * 8;
    const u16s* br = rbb + (size_t)(nt * 16 + lm) * Dn + hi * 8;
    v8f acc = vzero8();
    #pragma unroll
    for (int ks = 0; ks < Dn / 32; ++ks) {
      const int k0 = ks * 32;
      ABu aU, bU;
      *reinterpret_cast<uint4*>(&aU.u[0]) = *reinterpret_cast<const uint4*>(ar + k0);
      *reinterpret_cast<uint4*>(&aU.u[4]) = *reinterpret_cast<const uint4*>(ar + k0 + 16);
      *reinterpret_cast<uint4*>(&bU.u[0]) = *reinterpret_cast<const uint4*>(br + k0);
      *reinterpret_cast<uint4*>(&bU.u[4]) = *reinterpret_cast<const uint4*>(br + k0 + 16);
      acc = __builtin_amdgcn_wmma_f32_16x16x32_bf16(
          false, aU.v, false, bU.v, (short)0, acc, false, false);
    }
    #pragma unroll
    for (int r = 0; r < 8; ++r)
      Gs[(mt * 16 + hi * 8 + r) * 32 + nt * 16 + lm] = acc[r];
  }
  __syncthreads();

  const float inv_sqrt_d = 0.04419417382415922f;   // 1/sqrt(512)
  const float validf = s_valid;
  const int   nv     = (int)validf;

  // ---- iterative credibility: scores[q,k] = G[q,k] * Xb[k] * s ----
  for (int it = 0; it < ITERn; ++it) {
    const float s = inv_sqrt_d * scales_blk[it];
    if (t < Ln) {
      float a = 0.f;
      for (int k = 0; k < nv; ++k) a += Gs[t * 32 + k] * Xb[k];
      cred[t] = a * s / validf;
    }
    __syncthreads();
    if (t < Ln) {
      float z = b_o[it * Ln + t];
      for (int j = 0; j < Ln; ++j) z += cred[j] * W_o[it * Ln * Ln + t * Ln + j];
      Xb[t] = tanhf(expf(z));
    }
    __syncthreads();
  }

  // ---- evd_cred = tanh(exp(scores_fin @ W_c^T + b_c)) ----
  {
    const float sfin = inv_sqrt_d * scale_fin[0];
    if (t < Ln * Hn) {
      const int q = t >> 2, h = t & 3;
      float a = 0.f;
      for (int k = 0; k < Ln; ++k) a += Gs[q * 32 + k] * Xb[k] * W_c[h * Ln + k];
      const float e = tanhf(expf(sfin * a + b_c[h]));
      evd[t] = e;
      out2[((size_t)b * Ln + q) * (2 * Hn) + h] = e;
    }
  }

  // ---- Tl = tanh(leftW[b] + rightW[b]) ----
  {
    const float* lw = leftW  + (size_t)b * OUTn;
    const float* rw = rightW + (size_t)b * Ln * OUTn;
    for (int idx = t; idx < Ln * OUTn; idx += 128)
      Tl[idx] = tanhf(lw[idx & (OUTn - 1)] + rw[idx]);
  }
  __syncthreads();

  // ---- lin = Tl @ W2^T ----
  if (t < Ln * Hn) {
    const int l = t >> 2, h = t & 3;
    float a = 0.f;
    for (int o = 0; o < OUTn; ++o) a += Tl[l * OUTn + o] * W2[h * OUTn + o];
    lins[t] = a;
  }
  __syncthreads();

  // ---- masked softmax over L per head ----
  if (t < Hn) {
    float m = -3.4e38f;
    for (int l = 0; l < nv; ++l) m = fmaxf(m, lins[l * Hn + t]);
    float ssum = 0.f;
    for (int l = 0; l < nv; ++l) {
      const float e = __expf(lins[l * Hn + t] - m);
      aw[l * Hn + t] = e;
      ssum += e;
    }
    const float inv = 1.f / ssum;
    for (int l = 0;  l < nv; ++l) aw[l * Hn + t] *= inv;
    for (int l = nv; l < Ln; ++l) aw[l * Hn + t]  = 0.f;
  }
  __syncthreads();

  // ---- AOA GLU fusion ----
  if (t < Ln * Hn) {
    const int l = t >> 2, h = t & 3;
    float cat[8];
    #pragma unroll
    for (int j = 0; j < 4; ++j) { cat[j] = aw[l * 4 + j]; cat[4 + j] = evd[l * 4 + j]; }
    float ga = b_aoa[h], gb = b_aoa[h + 4];
    #pragma unroll
    for (int j = 0; j < 8; ++j) {
      ga += W_aoa[h * 8 + j] * cat[j];
      gb += W_aoa[(h + 4) * 8 + j] * cat[j];
    }
    const float v = ga * (1.f / (1.f + expf(-gb)));
    aw2s[t] = v;
    out2[((size_t)b * Ln + l) * (2 * Hn) + Hn + h] = v;
  }
  __syncthreads();

  // ---- attended[b,d,h] = sum_l right[b,l,d] * aw2[l,h] ----
  {
    float acc[4][4];
    #pragma unroll
    for (int i = 0; i < 4; ++i)
      #pragma unroll
      for (int h = 0; h < 4; ++h) acc[i][h] = 0.f;

    for (int l = 0; l < Ln; ++l) {
      const float w0 = aw2s[l * 4 + 0], w1 = aw2s[l * 4 + 1];
      const float w2v = aw2s[l * 4 + 2], w3 = aw2s[l * 4 + 3];
      #pragma unroll
      for (int i = 0; i < 4; ++i) {
        const float r = rb[l * Dn + t + i * 128];
        acc[i][0] += r * w0; acc[i][1] += r * w1;
        acc[i][2] += r * w2v; acc[i][3] += r * w3;
      }
    }
    #pragma unroll
    for (int i = 0; i < 4; ++i) {
      const int d = t + i * 128;
      #pragma unroll
      for (int h = 0; h < 4; ++h)
        out0[(size_t)b * Dn * Hn + (size_t)d * Hn + h] = acc[i][h];
    }
  }
}

// ---------------------------------------------------------------------------
extern "C" void kernel_launch(void* const* d_in, const int* in_sizes, int n_in,
                              void* d_out, int out_size, void* d_ws, size_t ws_size,
                              hipStream_t stream) {
  (void)in_sizes; (void)n_in; (void)out_size; (void)ws_size;
  const float* left       = (const float*)d_in[0];
  const float* right      = (const float*)d_in[1];
  const int*   mask       = (const int*)  d_in[2];
  const float* W1         = (const float*)d_in[3];
  const float* W2         = (const float*)d_in[4];
  const float* W_o        = (const float*)d_in[5];
  const float* b_o        = (const float*)d_in[6];
  const float* scales_blk = (const float*)d_in[7];
  const float* W_c        = (const float*)d_in[8];
  const float* b_c        = (const float*)d_in[9];
  const float* scale_fin  = (const float*)d_in[10];
  const float* W_aoa      = (const float*)d_in[11];
  const float* b_aoa      = (const float*)d_in[12];

  float* out  = (float*)d_out;
  float* out2 = out + (size_t)Bn * Dn * Hn;

  // workspace layout
  float* leftW   = (float*)d_ws;                       // B x 512 f32
  float* rightW  = leftW + (size_t)Bn * OUTn;          // (B*L) x 512 f32
  u16s*  W1bf    = (u16s*)(rightW + (size_t)Bn * Ln * OUTn);  // 512 x 1024 bf16
  u16s*  leftbf  = W1bf  + (size_t)OUTn * (Xn + Dn);   // 2048 x 512 bf16
  u16s*  rightbf = leftbf + (size_t)Bn * Xn;           // (B*L + 2) x 512 bf16

  dim3 blk128(128), blk256(256);

  // one-time f32 -> bf16 conversions
  {
    const int nW1 = OUTn * (Xn + Dn) / 8;       // 65536
    const int nL  = Bn * Xn / 8;                // 131072
    const int nR  = Bn * Ln * Dn / 8;           // 3932160
    cvt_f32_bf16<<<dim3((nW1 + 255) / 256), blk256, 0, stream>>>(W1, W1bf, nW1);
    cvt_f32_bf16<<<dim3((nL  + 255) / 256), blk256, 0, stream>>>(left, leftbf, nL);
    cvt_f32_bf16<<<dim3((nR  + 255) / 256), blk256, 0, stream>>>(right, rightbf, nR);
  }

  // leftW = left @ W1[:, :512]^T   (M = 2048)
  gemm128x64_bf16<Xn, Xn + Dn, OUTn><<<dim3(Bn / 128, OUTn / 64), blk128, 0, stream>>>(
      leftbf, W1bf, leftW);
  // rightW = right @ W1[:, 512:]^T (M = 61440, dominant GEMM); koff folded into W ptr
  gemm128x64_bf16<Dn, Xn + Dn, OUTn><<<dim3((Bn * Ln) / 128, OUTn / 64), blk128, 0, stream>>>(
      rightbf, W1bf + Xn, rightW);
  // fused per-batch pipeline
  fused_evd_attn<<<dim3(Bn), blk128, 0, stream>>>(
      right, rightbf, mask, W_o, b_o, scales_blk, W_c, b_c, scale_fin,
      W2, W_aoa, b_aoa, leftW, rightW, out, out2);
}
